// STGAT_cl_11759620456602
// MI455X (gfx1250) — compile-verified, matrix operand order checked by reading, not run on hardware
//
#include <hip/hip_runtime.h>

// ---------------------------------------------------------------------------
// STGAT forward for MI455X (gfx1250, wave32). Dense GEMMs via
// V_WMMA_F32_16X16X32_F16: 32x64 block tile, double-buffered LDS,
// global_load_b128 staging, ds_load_b128 fragment reads.
// ---------------------------------------------------------------------------

typedef __attribute__((ext_vector_type(16))) _Float16 v16h;
typedef __attribute__((ext_vector_type(8)))  _Float16 v8h;
typedef __attribute__((ext_vector_type(4)))  _Float16 v4h;
typedef __attribute__((ext_vector_type(8)))  float    v8f;

constexpr int CB   = 64;     // batch
constexpr int CL   = 12;     // timesteps
constexpr int CN   = 883;    // nodes
constexpr int CD   = 32;     // d_model
constexpr int CF   = 128;    // d_spat == d_temp
constexpr int CHID = 64;     // HID
constexpr int CHD  = 320;    // HIDDEN_DIM
constexpr int CHOR = 12;     // horizon
constexpr int LDH  = 896;    // padded node stride for hidden/z (%4==0 for b128)
constexpr int CES  = 16 * CN;  // 14128 spatial edges
constexpr int CET  = CL * CL;  // 144 temporal edges
constexpr int CLD  = CL * CD;  // 384
constexpr int CND  = CN * CD;  // 28256
constexpr long long PRED_ELEMS = (long long)CB * CHOR * CN;  // 678144

__device__ __forceinline__ float4 zero4() { float4 z; z.x = z.y = z.z = z.w = 0.f; return z; }

// ---------------------------------------------------------------------------
// Batched WMMA GEMM:  C[b] = act(A[b] * op(B[b]) + bias) (+ Res[b])
//   A row-major [M,K] (lda, %4==0).  B [K,N] (ldb, %4==0) or [N,K] if transB.
//   Loads of B columns are valid up to Nphys (%4==0, >= Nn); stores guarded
//   at Nn.  C stored at b*c_bs + m*c_rs + n*c_cs.  K % 32 == 0.
//   bias_mode: 0 none, 1 by n, 2 by m.
// Fragment layouts per CDNA5 ISA 7.12.2: each lane's 16-half A/B fragment is
// two contiguous 8-half chunks at column offsets {8*half, 16+8*half} of its
// matrix row -> two ds_load_b128 per operand.
// ---------------------------------------------------------------------------
__global__ __launch_bounds__(128) void k_wmma_gemm(
    const float* __restrict__ A, long long a_bs, int lda,
    const float* __restrict__ Bm, long long b_bs, int ldb, int transB,
    float* __restrict__ C, long long c_bs, long long c_rs, long long c_cs,
    const float* __restrict__ bias, int bias_mode,
    const float* __restrict__ Res, long long r_bs,
    int M, int Nn, int Nphys, int K, int relu)
{
    __shared__ __align__(16) _Float16 As[2][32][40];  // 80B row pitch
    __shared__ __align__(16) _Float16 Bs[2][64][40];  // stored as [n][k]
    const int b    = blockIdx.z;
    const int m0   = blockIdx.y * 32;
    const int n0   = blockIdx.x * 64;
    const float* Ab = A  + (long long)b * a_bs;
    const float* Bb = Bm + (long long)b * b_bs;
    const int tid  = threadIdx.x;
    const int wave = tid >> 5;
    const int lane = tid & 31;
    const int half = lane >> 4;
    const int lrow = lane & 15;

    float4 areg[2], breg[4];

    auto loadA = [&](int kt) {
#pragma unroll
        for (int j = 0; j < 2; ++j) {
            int r  = (tid >> 3) + j * 16;
            int gm = m0 + r;
            areg[j] = (gm < M)
                ? *reinterpret_cast<const float4*>(
                      Ab + (long long)gm * lda + (kt << 5) + ((tid & 7) << 2))
                : zero4();
        }
    };
    auto loadB = [&](int kt) {
        if (transB) {
#pragma unroll
            for (int j = 0; j < 4; ++j) {
                int nr = (tid >> 3) + j * 16;
                int gn = n0 + nr;
                breg[j] = (gn < Nphys)
                    ? *reinterpret_cast<const float4*>(
                          Bb + (long long)gn * ldb + (kt << 5) + ((tid & 7) << 2))
                    : zero4();
            }
        } else {
#pragma unroll
            for (int j = 0; j < 4; ++j) {
                int kr = (tid >> 4) + j * 8;
                int gn = n0 + ((tid & 15) << 2);
                breg[j] = (gn < Nphys)
                    ? *reinterpret_cast<const float4*>(
                          Bb + (long long)((kt << 5) + kr) * ldb + gn)
                    : zero4();
            }
        }
    };
    auto storeT = [&](int buf) {
#pragma unroll
        for (int j = 0; j < 2; ++j) {
            int r = (tid >> 3) + j * 16, c = (tid & 7) << 2;
            v4h h;
            h[0] = (_Float16)areg[j].x; h[1] = (_Float16)areg[j].y;
            h[2] = (_Float16)areg[j].z; h[3] = (_Float16)areg[j].w;
            *reinterpret_cast<v4h*>(&As[buf][r][c]) = h;
        }
        if (transB) {
#pragma unroll
            for (int j = 0; j < 4; ++j) {
                int nr = (tid >> 3) + j * 16, k4 = (tid & 7) << 2;
                v4h h;
                h[0] = (_Float16)breg[j].x; h[1] = (_Float16)breg[j].y;
                h[2] = (_Float16)breg[j].z; h[3] = (_Float16)breg[j].w;
                *reinterpret_cast<v4h*>(&Bs[buf][nr][k4]) = h;
            }
        } else {
#pragma unroll
            for (int j = 0; j < 4; ++j) {
                int kr = (tid >> 4) + j * 8, c = (tid & 15) << 2;
                Bs[buf][c + 0][kr] = (_Float16)breg[j].x;
                Bs[buf][c + 1][kr] = (_Float16)breg[j].y;
                Bs[buf][c + 2][kr] = (_Float16)breg[j].z;
                Bs[buf][c + 3][kr] = (_Float16)breg[j].w;
            }
        }
    };

    v8f acc0 = {}, acc1 = {};
    const int nk = K >> 5;
    loadA(0); loadB(0);
    storeT(0);
    __syncthreads();
    for (int kt = 0; kt < nk; ++kt) {
        const int buf = kt & 1;
        if (kt + 1 < nk) { loadA(kt + 1); loadB(kt + 1); }

        v8h bl = *reinterpret_cast<const v8h*>(&Bs[buf][(wave << 4) + lrow][half << 3]);
        v8h bh = *reinterpret_cast<const v8h*>(&Bs[buf][(wave << 4) + lrow][16 + (half << 3)]);
        v16h bf = __builtin_shufflevector(bl, bh, 0,1,2,3,4,5,6,7,8,9,10,11,12,13,14,15);
        v8h a0l = *reinterpret_cast<const v8h*>(&As[buf][lrow][half << 3]);
        v8h a0h = *reinterpret_cast<const v8h*>(&As[buf][lrow][16 + (half << 3)]);
        v16h af0 = __builtin_shufflevector(a0l, a0h, 0,1,2,3,4,5,6,7,8,9,10,11,12,13,14,15);
        v8h a1l = *reinterpret_cast<const v8h*>(&As[buf][16 + lrow][half << 3]);
        v8h a1h = *reinterpret_cast<const v8h*>(&As[buf][16 + lrow][16 + (half << 3)]);
        v16h af1 = __builtin_shufflevector(a1l, a1h, 0,1,2,3,4,5,6,7,8,9,10,11,12,13,14,15);

        acc0 = __builtin_amdgcn_wmma_f32_16x16x32_f16(false, af0, false, bf, (short)0, acc0, false, false);
        acc1 = __builtin_amdgcn_wmma_f32_16x16x32_f16(false, af1, false, bf, (short)0, acc1, false, false);

        if (kt + 1 < nk) {
            __syncthreads();
            storeT(buf ^ 1);
            __syncthreads();
        }
    }

    const int gn = n0 + (wave << 4) + lrow;
    if (gn < Nn) {
        const float bn_add = (bias_mode == 1) ? bias[gn] : 0.f;
#pragma unroll
        for (int sub = 0; sub < 2; ++sub) {
            v8f acc = sub ? acc1 : acc0;
#pragma unroll
            for (int v = 0; v < 8; ++v) {
                int gm = m0 + sub * 16 + v + (half << 3);
                if (gm < M) {
                    float val = acc[v] + bn_add;
                    if (bias_mode == 2) val += bias[gm];
                    if (relu) val = fmaxf(val, 0.f);
                    long long ci = (long long)gm * c_rs + (long long)gn * c_cs;
                    if (Res) val += Res[(long long)b * r_bs + ci];
                    C[(long long)b * c_bs + ci] = val;
                }
            }
        }
    }
}

// ---------------------------------------------------------------------------
// Small helper kernels
// ---------------------------------------------------------------------------
__global__ void k_fill_f(float* p, float v, long long n) {
    long long i = (long long)blockIdx.x * blockDim.x + threadIdx.x;
    if (i < n) p[i] = v;
}
__global__ void k_fill_i(int* p, int v, long long n) {
    long long i = (long long)blockIdx.x * blockDim.x + threadIdx.x;
    if (i < n) p[i] = v;
}

// X = hist[...,0:3] @ start_fc_w + b, laid out as res_x[b][n][l*32+d]
__global__ void k_start(const float* __restrict__ hist, const float* __restrict__ w,
                        const float* __restrict__ bias, float* __restrict__ res_x)
{
    long long idx = (long long)blockIdx.x * blockDim.x + threadIdx.x;
    const long long total = (long long)CB * CL * CN * CD;
    if (idx >= total) return;
    int d = (int)(idx & 31);
    long long t = idx >> 5;
    int n = (int)(t % CN); t /= CN;
    int l = (int)(t % CL);
    int b = (int)(t / CL);
    const float* hp = hist + (((long long)b * CL + l) * CN + n) * 3;
    float acc = bias[d] + hp[0] * w[d] + hp[1] * w[CD + d] + hp[2] * w[2 * CD + d];
    res_x[((long long)b * CN + n) * CLD + l * CD + d] = acc;
}

// graph_aug pass 1: x *= keep[node]; block-reduce sum into tok[b]
__global__ void k_aug_scale(float* __restrict__ x, const float* __restrict__ keep,
                            float* __restrict__ tok, int nnode, int dim)
{
    int b = blockIdx.y;
    long long idx = (long long)blockIdx.x * blockDim.x + threadIdx.x;
    long long total = (long long)nnode * dim;
    float v = 0.f;
    if (idx < total) {
        int n = (int)(idx / dim);
        float* xp = x + (long long)b * total + idx;
        v = *xp * keep[b * nnode + n];
        *xp = v;
    }
    __shared__ float red[256];
    red[threadIdx.x] = v;
    __syncthreads();
    for (int s = 128; s > 0; s >>= 1) {
        if (threadIdx.x < s) red[threadIdx.x] += red[threadIdx.x + s];
        __syncthreads();
    }
    if (threadIdx.x == 0) atomicAdd(&tok[b], red[0]);
}

// graph_aug pass 2: masked rows -> mean token
__global__ void k_aug_apply(float* __restrict__ x, const int* __restrict__ mask,
                            const float* __restrict__ tok, int nnode, int dim)
{
    int b = blockIdx.y;
    long long idx = (long long)blockIdx.x * blockDim.x + threadIdx.x;
    long long total = (long long)nnode * dim;
    if (idx >= total) return;
    int n = (int)(idx / dim);
    if (mask[b * nnode + n] > 0)
        x[(long long)b * total + idx] = tok[b] / (float)total;
}

// per-(node,head) attention source/dest scores
__global__ void k_attn_proj(const float* __restrict__ h, const float* __restrict__ a_src,
                            const float* __restrict__ a_dst, float* __restrict__ asrc,
                            float* __restrict__ adst, int nnode)
{
    long long idx = (long long)blockIdx.x * blockDim.x + threadIdx.x;
    long long total = (long long)CB * nnode * 4;
    if (idx >= total) return;
    int hh = (int)(idx & 3);
    long long t = idx >> 2;
    int n = (int)(t % nnode);
    int b = (int)(t / nnode);
    const float* hp = h + ((long long)b * nnode + n) * CF + hh * 32;
    float s1 = 0.f, s2 = 0.f;
    for (int d = 0; d < 32; ++d) {
        float v = hp[d];
        s1 += v * a_src[hh * 32 + d];
        s2 += v * a_dst[hh * 32 + d];
    }
    asrc[idx] = s1;
    adst[idx] = s2;
}

__device__ __forceinline__ int   fkey(float f) { int i = __float_as_int(f); return i >= 0 ? i : i ^ 0x7fffffff; }
__device__ __forceinline__ float funkey(int k) { return __int_as_float(k >= 0 ? k : k ^ 0x7fffffff); }
__device__ __forceinline__ float leaky02(float v) { return v > 0.f ? v : 0.2f * v; }

__global__ void k_edge_max(const int* __restrict__ edge, const float* __restrict__ asrc,
                           const float* __restrict__ adst, int* __restrict__ mint,
                           int E, int nnode)
{
    long long idx = (long long)blockIdx.x * blockDim.x + threadIdx.x;
    long long total = (long long)CB * E * 4;
    if (idx >= total) return;
    int hh = (int)(idx & 3);
    long long t = idx >> 2;
    int e = (int)(t % E);
    int b = (int)(t / E);
    int s = edge[(long long)b * 2 * E + e];
    int d = edge[(long long)b * 2 * E + E + e];
    float v = leaky02(asrc[((long long)b * nnode + s) * 4 + hh] +
                      adst[((long long)b * nnode + d) * 4 + hh]);
    atomicMax(&mint[((long long)b * nnode + d) * 4 + hh], fkey(v));
}

__global__ void k_edge_exp(const int* __restrict__ edge, const float* __restrict__ asrc,
                           const float* __restrict__ adst, const int* __restrict__ mint,
                           float* __restrict__ pbuf, float* __restrict__ den,
                           int E, int nnode)
{
    long long idx = (long long)blockIdx.x * blockDim.x + threadIdx.x;
    long long total = (long long)CB * E * 4;
    if (idx >= total) return;
    int hh = (int)(idx & 3);
    long long t = idx >> 2;
    int e = (int)(t % E);
    int b = (int)(t / E);
    int s = edge[(long long)b * 2 * E + e];
    int d = edge[(long long)b * 2 * E + E + e];
    float v = leaky02(asrc[((long long)b * nnode + s) * 4 + hh] +
                      adst[((long long)b * nnode + d) * 4 + hh]);
    float m = funkey(mint[((long long)b * nnode + d) * 4 + hh]);
    float p = __expf(v - m);
    pbuf[idx] = p;
    atomicAdd(&den[((long long)b * nnode + d) * 4 + hh], p);
}

__global__ void k_edge_aggr(const int* __restrict__ edge, const float* __restrict__ pbuf,
                            const float* __restrict__ den, const float* __restrict__ hmat,
                            float* __restrict__ out, int E, int nnode)
{
    long long idx = (long long)blockIdx.x * blockDim.x + threadIdx.x;
    long long total = (long long)CB * E * 4 * 32;
    if (idx >= total) return;
    int d32 = (int)(idx & 31);
    long long t = idx >> 5;
    int hh = (int)(t & 3); t >>= 2;
    int e = (int)(t % E);
    int b = (int)(t / E);
    int s = edge[(long long)b * 2 * E + e];
    int d = edge[(long long)b * 2 * E + E + e];
    float w = pbuf[((long long)b * E + e) * 4 + hh] /
              (den[((long long)b * nnode + d) * 4 + hh] + 1e-16f);
    atomicAdd(&out[((long long)b * nnode + d) * CF + hh * 32 + d32],
              w * hmat[((long long)b * nnode + s) * CF + hh * 32 + d32]);
}

// tempA[b][l][n*32+d] = spa_inv[b][n][l*32+d]
__global__ void k_pack_tempA(const float* __restrict__ spa_inv, float* __restrict__ tempA)
{
    int b = blockIdx.y;
    long long idx = (long long)blockIdx.x * blockDim.x + threadIdx.x;
    const long long total = (long long)CL * CND;
    if (idx >= total) return;
    int l = (int)(idx / CND);
    int r = (int)(idx % CND);
    int n = r / CD, d = r % CD;
    tempA[(long long)b * total + idx] =
        spa_inv[((long long)b * CN + n) * CLD + l * CD + d];
}

// flat[b][n][l*32+d] = temp_inv[b][l][n*32+d]
__global__ void k_pack_flat(const float* __restrict__ temp_inv, float* __restrict__ flat)
{
    int b = blockIdx.y;
    long long idx = (long long)blockIdx.x * blockDim.x + threadIdx.x;
    const long long total = (long long)CN * CLD;
    if (idx >= total) return;
    int n = (int)(idx / CLD);
    int r = (int)(idx % CLD);
    int l = r / CD, d = r % CD;
    flat[(long long)b * total + idx] =
        temp_inv[((long long)b * CL + l) * CND + n * CD + d];
}

// hidden[b][c][n] (stride LDH) = concat(ts_res, ts_temp, node_emb^T, T^T, D^T)
__global__ void k_build_hidden(const float* __restrict__ ts_res, const float* __restrict__ ts_temp,
                               const float* __restrict__ node_emb, const float* __restrict__ tid_emb,
                               const float* __restrict__ diw_emb, const float* __restrict__ hist,
                               float* __restrict__ hidden)
{
    int b = blockIdx.y;
    long long idx = (long long)blockIdx.x * blockDim.x + threadIdx.x;
    const long long total = (long long)CHD * LDH;
    if (idx >= total) return;
    int c = (int)(idx / LDH);
    int n = (int)(idx % LDH);
    float v = 0.f;
    if (n < CN) {
        if (c < 64)       v = ts_res[((long long)b * 64 + c) * CN + n];
        else if (c < 128) v = ts_temp[((long long)b * 64 + (c - 64)) * CN + n];
        else if (c < 192) v = node_emb[n * CHID + (c - 128)];
        else if (c < 256) {
            int ti = (int)hist[(((long long)b * CL + (CL - 1)) * CN + n) * 3 + 1];
            v = tid_emb[ti * CHID + (c - 192)];
        } else {
            int di = (int)hist[(((long long)b * CL + (CL - 1)) * CN + n) * 3 + 2];
            v = diw_emb[di * CHID + (c - 256)];
        }
    }
    hidden[(long long)b * total + idx] = v;
}

// per-node BN stats over (batch, channel): one block per node
__global__ void k_bn_stats(const float* __restrict__ hidden, float* __restrict__ mu,
                           float* __restrict__ rstd)
{
    int n = blockIdx.x;
    float s = 0.f, ss = 0.f;
    for (int i = threadIdx.x; i < CB * CHD; i += 256) {
        int b = i / CHD, c = i % CHD;
        float v = hidden[((long long)b * CHD + c) * LDH + n];
        s += v; ss += v * v;
    }
    __shared__ float rs[256], rq[256];
    rs[threadIdx.x] = s; rq[threadIdx.x] = ss;
    __syncthreads();
    for (int st = 128; st > 0; st >>= 1) {
        if (threadIdx.x < st) { rs[threadIdx.x] += rs[threadIdx.x + st];
                                rq[threadIdx.x] += rq[threadIdx.x + st]; }
        __syncthreads();
    }
    if (threadIdx.x == 0) {
        const float inv = 1.f / (float)(CB * CHD);
        float m = rs[0] * inv;
        float var = rq[0] * inv - m * m;
        mu[n] = m;
        rstd[n] = rsqrtf(var + 1e-5f);
    }
}

// output_rep[b][d] = sum_n proj_w[n]*BN(hidden[b][d][n]) + proj_b
__global__ void k_outrep(const float* __restrict__ hidden, const float* __restrict__ mu,
                         const float* __restrict__ rstd, const float* __restrict__ gamma,
                         const float* __restrict__ beta, const float* __restrict__ pw,
                         const float* __restrict__ pb, float* __restrict__ orep)
{
    int idx = blockIdx.x * blockDim.x + threadIdx.x;
    if (idx >= CB * CHD) return;
    int b = idx / CHD, d = idx % CHD;
    const float* hp = hidden + ((long long)b * CHD + d) * LDH;
    float acc = 0.f;
    for (int n = 0; n < CN; ++n) {
        float v = (hp[n] - mu[n]) * rstd[n] * gamma[n] + beta[n];
        acc += pw[n] * v;
    }
    orep[idx] = acc + pb[0];
}

// ---------------------------------------------------------------------------
// Host orchestration
// ---------------------------------------------------------------------------
static inline int cdiv(long long a, long long b) { return (int)((a + b - 1) / b); }

extern "C" void kernel_launch(void* const* d_in, const int* in_sizes, int n_in,
                              void* d_out, int out_size, void* d_ws, size_t ws_size,
                              hipStream_t stream)
{
    (void)in_sizes; (void)n_in; (void)out_size; (void)ws_size;
    const float* hist   = (const float*)d_in[0];
    const int*   e_spat = (const int*)  d_in[1];
    const int*   e_temp = (const int*)  d_in[2];
    const float* keep_s = (const float*)d_in[3];
    const float* keep_t = (const float*)d_in[4];
    const int*   mask_s = (const int*)  d_in[5];
    const int*   mask_t = (const int*)  d_in[6];
    const float* node_e = (const float*)d_in[7];
    const float* tid_e  = (const float*)d_in[8];
    const float* diw_e  = (const float*)d_in[9];
    const float* sfc_w  = (const float*)d_in[10];
    const float* sfc_b  = (const float*)d_in[11];
    const float* ts_w   = (const float*)d_in[12];
    const float* ts_b   = (const float*)d_in[13];
    const float* tsp_w  = (const float*)d_in[14];
    const float* tsp_b  = (const float*)d_in[15];
    const float* isp_w  = (const float*)d_in[16];
    const float* isp_b  = (const float*)d_in[17];
    const float* ttm_w  = (const float*)d_in[18];
    const float* ttm_b  = (const float*)d_in[19];
    const float* itm_w  = (const float*)d_in[20];
    const float* itm_b  = (const float*)d_in[21];
    const float* gs_w   = (const float*)d_in[22];
    const float* gs_as  = (const float*)d_in[23];
    const float* gs_ad  = (const float*)d_in[24];
    const float* gt_w   = (const float*)d_in[25];
    const float* gt_as  = (const float*)d_in[26];
    const float* gt_ad  = (const float*)d_in[27];
    const float* ew1    = (const float*)d_in[28];
    const float* eb1    = (const float*)d_in[29];
    const float* ew2    = (const float*)d_in[30];
    const float* eb2    = (const float*)d_in[31];
    const float* bng    = (const float*)d_in[32];
    const float* bnb    = (const float*)d_in[33];
    const float* pw     = (const float*)d_in[34];
    const float* pb     = (const float*)d_in[35];
    const float* rw     = (const float*)d_in[36];
    const float* rb     = (const float*)d_in[37];

    float* out  = (float*)d_out;
    float* pred = out;                       // [B,12,N,1]
    float* orep = out + PRED_ELEMS;          // [B,320]

    // ---- workspace layout (floats); bufA/bufB reused across phases ----
    float* ws = (float*)d_ws;
    size_t off = 0;
    auto alloc = [&](size_t n) { float* p = ws + off; off += n; return p; };
    float* bufA    = alloc((size_t)CB * CN * CLD);   // res_x -> tempA -> temp_flat
    float* bufB    = alloc((size_t)CB * CN * CLD);   // spa_inv -> temp_inv
    float* ts_res  = alloc((size_t)CB * 64 * CN);
    float* ts_temp = alloc((size_t)CB * 64 * CN);
    float* spa_inp = alloc((size_t)CB * CN * CF);
    float* h_s     = alloc((size_t)CB * CN * CF);
    float* spa_agg = alloc((size_t)CB * CN * CF);
    float* s_asrc  = alloc((size_t)CB * CN * 4);
    float* s_adst  = alloc((size_t)CB * CN * 4);
    int*   s_mint  = (int*)alloc((size_t)CB * CN * 4);
    float* s_den   = alloc((size_t)CB * CN * 4);
    float* s_p     = alloc((size_t)CB * CES * 4);
    float* tok     = alloc(64);
    float* temp_in = alloc((size_t)CB * CL * CF);
    float* h_t     = alloc((size_t)CB * CL * CF);
    float* tmp_agg = alloc((size_t)CB * CL * CF);
    float* t_asrc  = alloc((size_t)CB * CL * 4);
    float* t_adst  = alloc((size_t)CB * CL * 4);
    int*   t_mint  = (int*)alloc((size_t)CB * CL * 4);
    float* t_den   = alloc((size_t)CB * CL * 4);
    float* t_p     = alloc((size_t)CB * CET * 4);
    float* hidA    = alloc((size_t)CB * CHD * LDH);
    float* hidB    = alloc((size_t)CB * CHD * LDH);
    float* zbuf    = alloc((size_t)CB * CHD * LDH);
    float* mu      = alloc(1024);
    float* rstd    = alloc(1024);

    auto gemm = [&](const float* A, long long abs_, int lda,
                    const float* Bm, long long bbs, int ldb, int transB,
                    float* C, long long cbs, long long crs, long long ccs,
                    const float* bias, int bmode,
                    const float* res, long long rbs,
                    int M, int Nn, int Nphys, int K, int relu) {
        dim3 g(cdiv(Nn, 64), cdiv(M, 32), CB);
        k_wmma_gemm<<<g, 128, 0, stream>>>(A, abs_, lda, Bm, bbs, ldb, transB,
                                           C, cbs, crs, ccs, bias, bmode,
                                           res, rbs, M, Nn, Nphys, K, relu);
    };
    auto flat = [&](long long total) { return dim3(cdiv(total, 256)); };

    const long long abs_res = (long long)CN * CLD;   // 339072
    const long long abs_spa = (long long)CN * CF;    // 113024
    const long long abs_tmp = (long long)CL * CF;    // 1536
    const long long abs_hid = (long long)CHD * LDH;  // 286720

    // 1. start FC -> res_x (bufA)
    k_start<<<flat((long long)CB * abs_res), 256, 0, stream>>>(hist, sfc_w, sfc_b, bufA);

    // 2. ts_res[b][o][n] = res_x . ts_emb_w^T   (M=nodes, N=o, transB)
    gemm(bufA, abs_res, CLD, ts_w, 0, CLD, 1,
         ts_res, (long long)64 * CN, 1, CN, ts_b, 1, nullptr, 0, CN, 64, 64, CLD, 0);

    // 3. spa_inp = res_x @ trans_spat_w + b
    gemm(bufA, abs_res, CLD, tsp_w, 0, CF, 0,
         spa_inp, abs_spa, CF, 1, tsp_b, 1, nullptr, 0, CN, CF, CF, CLD, 0);

    // 4. spatial graph_aug (in place)
    k_fill_f<<<1, 64, 0, stream>>>(tok, 0.f, 64);
    {
        dim3 g(cdiv(abs_spa, 256), CB);
        k_aug_scale<<<g, 256, 0, stream>>>(spa_inp, keep_s, tok, CN, CF);
        k_aug_apply<<<g, 256, 0, stream>>>(spa_inp, mask_s, tok, CN, CF);
    }

    // 5. spatial GAT
    gemm(spa_inp, abs_spa, CF, gs_w, 0, CF, 0,
         h_s, abs_spa, CF, 1, nullptr, 0, nullptr, 0, CN, CF, CF, CF, 0);
    k_attn_proj<<<flat((long long)CB * CN * 4), 256, 0, stream>>>(h_s, gs_as, gs_ad, s_asrc, s_adst, CN);
    k_fill_i<<<flat((long long)CB * CN * 4), 256, 0, stream>>>(s_mint, (int)0x80000000, (long long)CB * CN * 4);
    k_edge_max<<<flat((long long)CB * CES * 4), 256, 0, stream>>>(e_spat, s_asrc, s_adst, s_mint, CES, CN);
    k_fill_f<<<flat((long long)CB * CN * 4), 256, 0, stream>>>(s_den, 0.f, (long long)CB * CN * 4);
    k_edge_exp<<<flat((long long)CB * CES * 4), 256, 0, stream>>>(e_spat, s_asrc, s_adst, s_mint, s_p, s_den, CES, CN);
    k_fill_f<<<flat((long long)CB * abs_spa), 256, 0, stream>>>(spa_agg, 0.f, (long long)CB * abs_spa);
    k_edge_aggr<<<flat((long long)CB * CES * 128), 256, 0, stream>>>(e_spat, s_p, s_den, h_s, spa_agg, CES, CN);

    // 6. spa_inv = spa_agg @ inv_spat_w + b   -> bufB
    gemm(spa_agg, abs_spa, CF, isp_w, 0, CLD, 0,
         bufB, abs_res, CLD, 1, isp_b, 1, nullptr, 0, CN, CLD, CLD, CF, 0);

    // 7. repack L<->N  -> tempA (bufA, res_x dead)
    { dim3 g(cdiv(abs_res, 256), CB);
      k_pack_tempA<<<g, 256, 0, stream>>>(bufB, bufA); }

    // 8. temp_in = tempA @ trans_temp_w + b   [12 x 128], K=28256
    gemm(bufA, abs_res, CND, ttm_w, 0, CF, 0,
         temp_in, abs_tmp, CF, 1, ttm_b, 1, nullptr, 0, CL, CF, CF, CND, 0);

    // 9. temporal graph_aug
    k_fill_f<<<1, 64, 0, stream>>>(tok, 0.f, 64);
    {
        dim3 g(cdiv(abs_tmp, 256), CB);
        k_aug_scale<<<g, 256, 0, stream>>>(temp_in, keep_t, tok, CL, CF);
        k_aug_apply<<<g, 256, 0, stream>>>(temp_in, mask_t, tok, CL, CF);
    }

    // 10. temporal GAT
    gemm(temp_in, abs_tmp, CF, gt_w, 0, CF, 0,
         h_t, abs_tmp, CF, 1, nullptr, 0, nullptr, 0, CL, CF, CF, CF, 0);
    k_attn_proj<<<flat((long long)CB * CL * 4), 256, 0, stream>>>(h_t, gt_as, gt_ad, t_asrc, t_adst, CL);
    k_fill_i<<<flat((long long)CB * CL * 4), 256, 0, stream>>>(t_mint, (int)0x80000000, (long long)CB * CL * 4);
    k_edge_max<<<flat((long long)CB * CET * 4), 256, 0, stream>>>(e_temp, t_asrc, t_adst, t_mint, CET, CL);
    k_fill_f<<<flat((long long)CB * CL * 4), 256, 0, stream>>>(t_den, 0.f, (long long)CB * CL * 4);
    k_edge_exp<<<flat((long long)CB * CET * 4), 256, 0, stream>>>(e_temp, t_asrc, t_adst, t_mint, t_p, t_den, CET, CL);
    k_fill_f<<<flat((long long)CB * abs_tmp), 256, 0, stream>>>(tmp_agg, 0.f, (long long)CB * abs_tmp);
    k_edge_aggr<<<flat((long long)CB * CET * 128), 256, 0, stream>>>(e_temp, t_p, t_den, h_t, tmp_agg, CET, CL);

    // 11. temp_inv = tmp_agg @ inv_temp_w + b  [12 x 28256] -> bufB
    gemm(tmp_agg, abs_tmp, CF, itm_w, 0, CND, 0,
         bufB, abs_res, CND, 1, itm_b, 1, nullptr, 0, CL, CND, CND, CF, 0);

    // 12. repack N<->L -> temp_flat (bufA, tempA dead)
    { dim3 g(cdiv(abs_res, 256), CB);
      k_pack_flat<<<g, 256, 0, stream>>>(bufB, bufA); }

    // 13. ts_temp[b][o][n]
    gemm(bufA, abs_res, CLD, ts_w, 0, CLD, 1,
         ts_temp, (long long)64 * CN, 1, CN, ts_b, 1, nullptr, 0, CN, 64, 64, CLD, 0);

    // 14. hidden concat -> hidA (padded stride LDH)
    { dim3 g(cdiv(abs_hid, 256), CB);
      k_build_hidden<<<g, 256, 0, stream>>>(ts_res, ts_temp, node_e, tid_e, diw_e, hist, hidA); }

    // 15. encoder: 3x { z = relu(W1 h + b1); h = h + W2 z + b2 }
    float* cur = hidA;
    float* nxt = hidB;
    for (int i = 0; i < 3; ++i) {
        gemm(ew1 + (size_t)i * CHD * CHD, 0, CHD, cur, abs_hid, LDH, 0,
             zbuf, abs_hid, LDH, 1, eb1 + i * CHD, 2, nullptr, 0, CHD, CN, LDH, CHD, 1);
        gemm(ew2 + (size_t)i * CHD * CHD, 0, CHD, zbuf, abs_hid, LDH, 0,
             nxt, abs_hid, LDH, 1, eb2 + i * CHD, 2, cur, abs_hid, CHD, CN, LDH, CHD, 0);
        float* t = cur; cur = nxt; nxt = t;
    }

    // 16. prediction = reg_w @ hidden + reg_b  -> d_out[0 .. B*12*N)
    gemm(rw, 0, CHD, cur, abs_hid, LDH, 0,
         pred, (long long)CHOR * CN, CN, 1, rb, 2, nullptr, 0, CHOR, CN, LDH, CHD, 0);

    // 17. BN stats + output_rep
    k_bn_stats<<<CN, 256, 0, stream>>>(cur, mu, rstd);
    k_outrep<<<cdiv(CB * CHD, 256), 256, 0, stream>>>(cur, mu, rstd, bng, bnb, pw, pb, orep);
}
